// Model_41532333753126
// MI455X (gfx1250) — matrix operand order repaired and checked by reading, not compiled, not run on hardware
//
#include <hip/hip_runtime.h>
#include <hip/hip_bf16.h>

// ---------------------------------------------------------------------------
// Types for CDNA5 WMMA
// ---------------------------------------------------------------------------
typedef __attribute__((ext_vector_type(16))) __bf16 v16bf;
typedef __attribute__((ext_vector_type(8)))  float  v8f;

union AFrag { v16bf v; unsigned u[8]; };

#define ACT_NONE     0
#define ACT_GELU     1
#define ACT_RELU     2
#define ACT_RELU_RES 3

__device__ __forceinline__ unsigned short f2bf(float f) {
  unsigned u = __float_as_uint(f);
  u += 0x7FFFu + ((u >> 16) & 1u);          // round-to-nearest-even
  return (unsigned short)(u >> 16);
}

__device__ __forceinline__ float gelu_exact(float x) {
  return 0.5f * x * (1.0f + erff(x * 0.70710678118654752440f));
}

// ---------------------------------------------------------------------------
// f32 -> bf16 conversion
// ---------------------------------------------------------------------------
__global__ void cvt_f32_bf16(const float* __restrict__ src,
                             unsigned short* __restrict__ dst, int n) {
  int i = blockIdx.x * blockDim.x + threadIdx.x;
  if (i < n) dst[i] = f2bf(src[i]);
}

__global__ void zero_f32(float* __restrict__ p, long long n) {
  long long i = (long long)blockIdx.x * blockDim.x + threadIdx.x;
  if (i < n) p[i] = 0.0f;
}

__global__ void zero_i32(int* __restrict__ p, int n) {
  int i = blockIdx.x * blockDim.x + threadIdx.x;
  if (i < n) p[i] = 0;
}

// ---------------------------------------------------------------------------
// bf16 GEMM with v_wmma_f32_16x16x32_bf16.
//   C(MxN) = act(A(MxK) * B(KxN) + bias [, + residual])
// Block tile 128x64, 8 waves, each wave a 32x32 tile (2x2 WMMA frags).
// Double-buffered LDS; A tiles staged with global_load_async_to_lds_b128
// (ASYNCcnt pipeline), B tiles transposed through VGPRs into n-major LDS.
// Requires M%128==0, N%64==0, K%32==0 (true for every GEMM in this model).
// ---------------------------------------------------------------------------
__launch_bounds__(256)
__global__ void gemm_bf16_wmma(const unsigned short* __restrict__ A,
                               const unsigned short* __restrict__ B,
                               const float* __restrict__ bias,
                               const float* __restrict__ residual,
                               float* __restrict__ C,
                               unsigned short* __restrict__ Cb,
                               int M, int N, int K, int act) {
  __shared__ __align__(16) unsigned short lA[2][128 * 40];  // 128 rows x (32+8 pad)
  __shared__ __align__(16) unsigned short lB[2][64 * 40];   // 64 cols (n-major) x (32+8 pad)

  const int tid  = threadIdx.x;
  const int lane = tid & 31;
  const int wave = tid >> 5;
  const int wm   = wave >> 1;   // 0..3 : M sub-tile
  const int wn   = wave & 1;    // 0..1 : N sub-tile
  const int ll   = lane & 15;
  const int lh   = lane >> 4;

  const int rowBase = blockIdx.y * 128;
  const int colBase = blockIdx.x * 64;

  v8f acc[2][2];
#pragma unroll
  for (int i = 0; i < 2; ++i)
#pragma unroll
    for (int j = 0; j < 2; ++j)
#pragma unroll
      for (int e = 0; e < 8; ++e) acc[i][j][e] = 0.0f;

  const int arow  = tid >> 1;   // 0..127
  const int ahalf = tid & 1;    // which 16-element half of the 32-wide k slab
  const int bk    = tid >> 3;   // 0..31
  const int bseg  = tid & 7;    // 8-col segment

  // Issue one 32-wide K slab into stage st: A via async copy, B via transpose.
  auto issueTile = [&](int k0, int st) {
    const unsigned short* srcA = A + (size_t)(rowBase + arow) * K + (k0 + ahalf * 16);
    unsigned ldsA = (unsigned)(size_t)&lA[st][arow * 40 + ahalf * 16];
    asm volatile("global_load_async_to_lds_b128 %0, %1, off"
                 :: "v"(ldsA), "v"((unsigned long long)(size_t)srcA) : "memory");
    asm volatile("global_load_async_to_lds_b128 %0, %1, off"
                 :: "v"(ldsA + 16u), "v"((unsigned long long)(size_t)(srcA + 8)) : "memory");

    const unsigned short* srcB = B + (size_t)(k0 + bk) * N + (colBase + bseg * 8);
    union { uint4 v; unsigned short s[8]; } tb;
    tb.v = *(const uint4*)(srcB);
#pragma unroll
    for (int j = 0; j < 8; ++j) lB[st][(bseg * 8 + j) * 40 + bk] = tb.s[j];
  };

  const int nk = K >> 5;
  issueTile(0, 0);

  for (int i = 0; i < nk; ++i) {
    const int s = i & 1;
    if (i + 1 < nk) {
      issueTile((i + 1) << 5, 1 - s);
      asm volatile("s_wait_asynccnt 0x2" ::: "memory");  // stage s A landed
    } else {
      asm volatile("s_wait_asynccnt 0x0" ::: "memory");
    }
    __syncthreads();                                     // stage s visible to all

    // Fragment loads per ISA 16-bit A/B layouts (lane = row/col, lh = K half)
    AFrag a[2], b[2];
    const unsigned* pA = (const unsigned*)&lA[s][0];
    const unsigned* pB = (const unsigned*)&lB[s][0];
#pragma unroll
    for (int mi = 0; mi < 2; ++mi) {
      const unsigned* p = pA + (wm * 32 + mi * 16 + ll) * 20 + lh * 4;
#pragma unroll
      for (int q = 0; q < 4; ++q) { a[mi].u[q] = p[q]; a[mi].u[4 + q] = p[8 + q]; }
    }
#pragma unroll
    for (int ni = 0; ni < 2; ++ni) {
      const unsigned* p = pB + (wn * 32 + ni * 16 + ll) * 20 + lh * 4;
#pragma unroll
      for (int q = 0; q < 4; ++q) { b[ni].u[q] = p[q]; b[ni].u[4 + q] = p[8 + q]; }
    }

#pragma unroll
    for (int mi = 0; mi < 2; ++mi)
#pragma unroll
      for (int ni = 0; ni < 2; ++ni)
        acc[mi][ni] = __builtin_amdgcn_wmma_f32_16x16x32_bf16(
            false, a[mi].v, false, b[ni].v, (short)0, acc[mi][ni], false, false);

    __syncthreads();   // all reads of stage s done before it is refilled
  }

  // Epilogue: bias + activation (+ residual), f32 and/or bf16 output
#pragma unroll
  for (int mi = 0; mi < 2; ++mi) {
#pragma unroll
    for (int ni = 0; ni < 2; ++ni) {
      int n = colBase + wn * 32 + ni * 16 + ll;
      float bv = bias ? bias[n] : 0.0f;
#pragma unroll
      for (int v = 0; v < 8; ++v) {
        int m = rowBase + wm * 32 + mi * 16 + lh * 8 + v;
        float x = acc[mi][ni][v] + bv;
        if (act == ACT_GELU)          x = gelu_exact(x);
        else if (act == ACT_RELU)     x = fmaxf(x, 0.0f);
        else if (act == ACT_RELU_RES) x = residual[(size_t)m * N + n] + fmaxf(x, 0.0f);
        size_t o = (size_t)m * N + n;
        if (C)  C[o]  = x;
        if (Cb) Cb[o] = f2bf(x);
      }
    }
  }
}

// ---------------------------------------------------------------------------
// Vector quantizer: one wave per flattened (b,l) vector (16384 total).
// zp layout: [(b*64 + c)*512 + l]  (encoder output, pre-transpose)
// ---------------------------------------------------------------------------
__launch_bounds__(256)
__global__ void vq_kernel(const float* __restrict__ zp,
                          const float* __restrict__ cb,        // (512, 64)
                          float* __restrict__ enc,             // (16384, 512), pre-zeroed
                          float* __restrict__ qd,              // (2048, 512) output
                          unsigned short* __restrict__ qdb,    // bf16 copy for decoder GEMM
                          float* __restrict__ dmin,
                          int* __restrict__ hist) {
  __shared__ float sh[8][64];
  const int tid = threadIdx.x, lane = tid & 31, wave = tid >> 5;
  const int r = blockIdx.x * 8 + wave;   // 0..16383
  const int b = r >> 9, l = r & 511;

  // gather the 64-dim vector (strided over channel rows)
  sh[wave][lane]      = zp[(size_t)(b * 64 + lane) * 512 + l];
  sh[wave][lane + 32] = zp[(size_t)(b * 64 + lane + 32) * 512 + l];
  __syncthreads();

  float best = 3.4e38f; int bidx = 0;
  for (int kk = 0; kk < 16; ++kk) {
    int k = lane * 16 + kk;                       // lane L owns codes [16L,16L+16)
    const float* c = cb + k * 64;
    float d = 0.0f;
#pragma unroll 8
    for (int cc = 0; cc < 64; ++cc) {
      float t = sh[wave][cc] - c[cc];
      d = fmaf(t, t, d);
    }
    if (d < best) { best = d; bidx = k; }         // ascending k -> first-min kept
  }
  // butterfly argmin, tie-break = lowest index (matches jnp.argmin)
  for (int off = 16; off >= 1; off >>= 1) {
    float od = __shfl_xor(best, off, 32);
    int   oi = __shfl_xor(bidx, off, 32);
    if (od < best || (od == best && oi < bidx)) { best = od; bidx = oi; }
  }

  if (lane == 0) {
    atomicAdd(&hist[bidx], 1);
    dmin[r] = best;                                // sum over dims of (z - q)^2
    enc[(size_t)r * 512 + bidx] = 1.0f;
  }
  float q0 = cb[bidx * 64 + lane];
  float q1 = cb[bidx * 64 + lane + 32];
  size_t o0 = (size_t)(b * 64 + lane) * 512 + l;
  size_t o1 = (size_t)(b * 64 + lane + 32) * 512 + l;
  qd[o0] = q0;  qd[o1] = q1;            // STE forward value == q, back in (B,C,K) layout
  qdb[o0] = f2bf(q0); qdb[o1] = f2bf(q1);
}

// ---------------------------------------------------------------------------
// Deterministic reduction: loss = 1.25 * MSE, perplexity from histogram
// ---------------------------------------------------------------------------
__launch_bounds__(256)
__global__ void finalize_kernel(const float* __restrict__ dmin,
                                const int* __restrict__ hist,
                                float* __restrict__ loss_out,
                                float* __restrict__ perp_out) {
  __shared__ float sA[256], sB[256];
  int tid = threadIdx.x;
  float s = 0.0f, p = 0.0f;
  for (int i = tid; i < 16384; i += 256) s += dmin[i];
  for (int i = tid; i < 512; i += 256) {
    float avg = (float)hist[i] * (1.0f / 16384.0f);
    p += avg * logf(avg + 1e-10f);
  }
  sA[tid] = s; sB[tid] = p;
  __syncthreads();
  for (int st = 128; st > 0; st >>= 1) {
    if (tid < st) { sA[tid] += sA[tid + st]; sB[tid] += sB[tid + st]; }
    __syncthreads();
  }
  if (tid == 0) {
    float mse = sA[0] / (16384.0f * 64.0f);  // q_loss == e_loss numerically
    *loss_out = 1.25f * mse;                 // q_loss + 0.25 * e_loss
    *perp_out = expf(-sB[0]);
  }
}

// ---------------------------------------------------------------------------
// Host driver
// ---------------------------------------------------------------------------
extern "C" void kernel_launch(void* const* d_in, const int* in_sizes, int n_in,
                              void* d_out, int out_size, void* d_ws, size_t ws_size,
                              hipStream_t stream) {
  (void)in_sizes; (void)n_in; (void)out_size; (void)ws_size;
  const float* x    = (const float*)d_in[0];
  const float* We1  = (const float*)d_in[1];
  const float* be1  = (const float*)d_in[2];
  const float* We2  = (const float*)d_in[3];
  const float* be2  = (const float*)d_in[4];
  const float* Wre1 = (const float*)d_in[5];
  const float* bre1 = (const float*)d_in[6];
  const float* Wre2 = (const float*)d_in[7];
  const float* bre2 = (const float*)d_in[8];
  const float* Wpre = (const float*)d_in[9];
  const float* bpre = (const float*)d_in[10];
  const float* cbk  = (const float*)d_in[11];
  const float* Wd0  = (const float*)d_in[12];
  const float* bd0  = (const float*)d_in[13];
  const float* Wrd1 = (const float*)d_in[14];
  const float* brd1 = (const float*)d_in[15];
  const float* Wrd2 = (const float*)d_in[16];
  const float* brd2 = (const float*)d_in[17];
  const float* Wd1  = (const float*)d_in[18];
  const float* bd1  = (const float*)d_in[19];
  const float* Wd2  = (const float*)d_in[20];
  const float* bd2  = (const float*)d_in[21];

  const int M = 2048;                // B * C_IN

  // ---- output layout: loss | x_recon | perplexity | enc | qd ----
  float* out      = (float*)d_out;
  float* out_loss = out;
  float* out_xrec = out + 1;                                   // 2048*16384
  float* out_perp = out + 1 + (long long)2048 * 16384;
  float* out_enc  = out_perp + 1;                              // 16384*512
  float* out_qd   = out_enc + (long long)16384 * 512;          // 2048*512

  // ---- workspace carving (256B aligned) ----
  char* base = (char*)d_ws;
  size_t off = 0;
  auto carve = [&](size_t bytes) -> char* {
    off = (off + 255) & ~(size_t)255;
    char* p = base + off; off += bytes; return p;
  };
  unsigned short* xb   = (unsigned short*)carve((size_t)2048 * 16384 * 2);
  unsigned short* wbig = (unsigned short*)carve((size_t)16384 * 2048 * 2);  // We1b / Wd2b
  unsigned short* wtmp = (unsigned short*)carve((size_t)524288 * 2);        // small weights
  unsigned short* z1b  = (unsigned short*)carve((size_t)2048 * 2048 * 2);
  float*          z2f  = (float*)carve((size_t)2048 * 256 * 4);
  unsigned short* z2b  = (unsigned short*)carve((size_t)2048 * 256 * 2);
  unsigned short* r1b  = (unsigned short*)carve((size_t)2048 * 64 * 2);
  unsigned short* z3b  = (unsigned short*)carve((size_t)2048 * 256 * 2);
  float*          zpf  = (float*)carve((size_t)2048 * 512 * 4);
  unsigned short* qdb  = (unsigned short*)carve((size_t)2048 * 512 * 2);
  float*          h0f  = (float*)carve((size_t)2048 * 256 * 4);
  unsigned short* h0b  = (unsigned short*)carve((size_t)2048 * 256 * 2);
  unsigned short* rr1b = (unsigned short*)carve((size_t)2048 * 64 * 2);
  unsigned short* h1b  = (unsigned short*)carve((size_t)2048 * 256 * 2);
  unsigned short* h2b  = (unsigned short*)carve((size_t)2048 * 2048 * 2);
  float*          dmin = (float*)carve((size_t)16384 * 4);
  int*            hist = (int*)carve((size_t)512 * 4);

  auto cvt = [&](const float* s, unsigned short* d, int n) {
    cvt_f32_bf16<<<(n + 255) / 256, 256, 0, stream>>>(s, d, n);
  };
  auto gemm = [&](const unsigned short* A, const unsigned short* Bm, const float* bias,
                  const float* res, float* Cf, unsigned short* Cb2,
                  int Mm, int Nn, int Kk, int act) {
    dim3 grid(Nn / 64, Mm / 128);
    gemm_bf16_wmma<<<grid, 256, 0, stream>>>(A, Bm, bias, res, Cf, Cb2, Mm, Nn, Kk, act);
  };

  // ---- Encoder ----
  cvt(x, xb, 2048 * 16384);
  cvt(We1, wbig, 16384 * 2048);
  gemm(xb, wbig, be1, nullptr, nullptr, z1b, M, 2048, 16384, ACT_GELU);
  cvt(We2, wtmp, 2048 * 256);
  gemm(z1b, wtmp, be2, nullptr, z2f, z2b, M, 256, 2048, ACT_GELU);
  cvt(Wre1, wtmp, 256 * 64);
  gemm(z2b, wtmp, bre1, nullptr, nullptr, r1b, M, 64, 256, ACT_RELU);
  cvt(Wre2, wtmp, 64 * 256);
  gemm(r1b, wtmp, bre2, z2f, nullptr, z3b, M, 256, 64, ACT_RELU_RES);
  cvt(Wpre, wtmp, 256 * 512);
  gemm(z3b, wtmp, bpre, nullptr, zpf, nullptr, M, 512, 256, ACT_NONE);

  // ---- Vector quantizer ----
  zero_f32<<<(int)(((long long)16384 * 512 + 255) / 256), 256, 0, stream>>>(
      out_enc, (long long)16384 * 512);
  zero_i32<<<2, 256, 0, stream>>>(hist, 512);
  vq_kernel<<<16384 / 8, 256, 0, stream>>>(zpf, cbk, out_enc, out_qd, qdb, dmin, hist);
  finalize_kernel<<<1, 256, 0, stream>>>(dmin, hist, out_loss, out_perp);

  // ---- Decoder ----
  cvt(Wd0, wtmp, 512 * 256);
  gemm(qdb, wtmp, bd0, nullptr, h0f, h0b, M, 256, 512, ACT_GELU);
  cvt(Wrd1, wtmp, 256 * 64);
  gemm(h0b, wtmp, brd1, nullptr, nullptr, rr1b, M, 64, 256, ACT_RELU);
  cvt(Wrd2, wtmp, 64 * 256);
  gemm(rr1b, wtmp, brd2, h0f, nullptr, h1b, M, 256, 64, ACT_RELU_RES);
  cvt(Wd1, wtmp, 256 * 2048);
  gemm(h1b, wtmp, bd1, nullptr, nullptr, h2b, M, 2048, 256, ACT_GELU);
  cvt(Wd2, wbig, 2048 * 16384);
  gemm(h2b, wbig, bd2, nullptr, out_xrec, nullptr, M, 16384, 2048, ACT_GELU);
}
